// VideoEmbedding_80178449482037
// MI455X (gfx1250) — matrix operand order repaired and checked by reading, not compiled, not run on hardware
//
#include <hip/hip_runtime.h>
#include <hip/hip_bf16.h>

typedef _Float16 v8h  __attribute__((ext_vector_type(8)));
typedef _Float16 v16h __attribute__((ext_vector_type(16)));
typedef float    v8f  __attribute__((ext_vector_type(8)));

#define NFRAMES   40      // B*F = 2*20
#define GROUPS    9       // 144 patches / 16 per group
#define NB_HALFS  65536   // 16 tiles * 8 kblocks * 32 lanes * 16 halfs

// ---------------------------------------------------------------------------
// Kernel 1: pack emb_weight [256 bins x 256 chans] f32 into hi/lo f16 in the
// wave32 B-operand order chosen here (contiguous k, as A) for v_wmma_f32_16x16x32_f16:
//   flat id = ((n_tile*8 + kblock)*32 + lane)*16 + slot
//   col  = n_tile*16 + (lane & 15)
//   krow = kblock*32 + ((lane>=16) ? 16 : 0) + slot
// ---------------------------------------------------------------------------
__global__ void prep_emb_kernel(const float* __restrict__ emb,
                                _Float16* __restrict__ Bhi,
                                _Float16* __restrict__ Blo) {
    int id = blockIdx.x * 256 + threadIdx.x;        // 0..65535
    int s    = id & 15;
    int lane = (id >> 4) & 31;
    int kk   = (id >> 9) & 7;
    int n_t  = (id >> 12) & 15;
    int col  = n_t * 16 + (lane & 15);
    int krow = kk * 32 + ((lane >> 4) & 1) * 16 + s;
    float v = emb[krow * 256 + col];
    _Float16 h = (_Float16)v;
    _Float16 l = (_Float16)(v - (float)h);          // residual: emb = h + l
    Bhi[id] = h;
    Blo[id] = l;
}

// ---------------------------------------------------------------------------
// Kernel 3: cls slot = temporal embedding only
// ---------------------------------------------------------------------------
__global__ void cls_fill_kernel(const float* __restrict__ temp,
                                float* __restrict__ out) {
    int frame = blockIdx.x;                          // 0..39
    int c = threadIdx.x;                             // 0..255
    out[(size_t)frame * 145 * 256 + c] = temp[(frame % 20) * 256 + c];
}

// ---------------------------------------------------------------------------
// Kernel 2: per (frame, 16-patch group): histogram -> WMMA GEMM -> epilogue
// ---------------------------------------------------------------------------
__global__ void __launch_bounds__(256)
video_embed_kernel(const float* __restrict__ x,
                   const _Float16* __restrict__ Bhi,
                   const _Float16* __restrict__ Blo,
                   const float* __restrict__ pos,
                   const float* __restrict__ temp,
                   float* __restrict__ out) {
    __shared__ unsigned  hist[16 * 256];   // 16 KB: per-patch 256-bin counts
    __shared__ _Float16  cntf[16 * 256];   // 8 KB: counts as f16 (A operand)

    const int tid   = threadIdx.x;
    const int wg    = blockIdx.x;          // 0..359
    const int frame = wg / GROUPS;         // 0..39  (== b*20 + f)
    const int g     = wg % GROUPS;         // patch group 0..8
    const int f     = frame % 20;
    const float* xf = x + (size_t)frame * 240 * 240;

    // ---- Phase A: histogram ----
    for (int i = tid; i < 16 * 256; i += 256) hist[i] = 0u;
    __syncthreads();

    // 16 patches * 400 pixels = 6400 pixels, 25 per thread, coalesced in q
    for (int t = 0; t < 25; ++t) {
        int q  = t * 256 + tid;            // 0..6399
        int pl = q / 400;                  // patch within group
        int r  = q % 400;                  // pixel within patch
        int pi = r / 20;                   // row in patch
        int pj = r % 20;                   // col in patch
        int p  = g * 16 + pl;              // patch 0..143 (row-major y,x)
        int py = p / 12, px = p % 12;
        float v = xf[(py * 20 + pi) * 240 + px * 20 + pj];
        int idx = (int)(v * 255.0f);       // truncation, matches .int()
        idx = idx < 0 ? 0 : (idx > 255 ? 255 : idx);
        atomicAdd(&hist[pl * 256 + idx], 1u);
    }
    __syncthreads();

    for (int i = tid; i < 16 * 256; i += 256)
        cntf[i] = (_Float16)(float)hist[i];   // counts <= 400: exact in f16
    __syncthreads();

    // ---- Phase B: GEMM  C[16x256] = counts[16x256] x emb[256x256] ----
    const int lane = tid & 31;
    const int wave = tid >> 5;             // 0..7
    const int hi16 = (lane >> 4) & 1;      // lane group
    const int mrow = lane & 15;            // A row for this lane
    const float scale = 1.0f / 400.0f;

    for (int tt = 0; tt < 2; ++tt) {
        const int n_t = wave * 2 + tt;     // output col tile 0..15
        const int n0  = n_t * 16;
        v8f acc = {};

        for (int kk = 0; kk < 8; ++kk) {   // K = 256 in blocks of 32
            // A operand: 16x32 f16, contiguous k order (as B pack) — lanes 0-15: K {0..15},
            // lanes 16-31: K {16..31}, row M = lane&15
            int abase = mrow * 256 + kk * 32 + hi16 * 16;
            v8h alo = *(const v8h*)&cntf[abase];
            v8h ahi = *(const v8h*)&cntf[abase + 8];
            v16h a = __builtin_shufflevector(alo, ahi,
                        0,1,2,3,4,5,6,7,8,9,10,11,12,13,14,15);

            const size_t bofs = ((size_t)(n_t * 8 + kk) * 32 + lane) * 16;

            v8h bh0 = *(const v8h*)(Bhi + bofs);
            v8h bh1 = *(const v8h*)(Bhi + bofs + 8);
            v16h bh = __builtin_shufflevector(bh0, bh1,
                        0,1,2,3,4,5,6,7,8,9,10,11,12,13,14,15);
            acc = __builtin_amdgcn_wmma_f32_16x16x32_f16(
                    false, a, false, bh, (short)0, acc, false, false);

            v8h bl0 = *(const v8h*)(Blo + bofs);
            v8h bl1 = *(const v8h*)(Blo + bofs + 8);
            v16h bl = __builtin_shufflevector(bl0, bl1,
                        0,1,2,3,4,5,6,7,8,9,10,11,12,13,14,15);
            acc = __builtin_amdgcn_wmma_f32_16x16x32_f16(
                    false, a, false, bl, (short)0, acc, false, false);
        }

        // ---- Epilogue: D layout — VGPR r, lanes 0-15: M=r, N=lane;
        //                lanes 16-31: M=r+8, N=lane-16 ----
        const int col = n0 + mrow;
        #pragma unroll
        for (int r = 0; r < 8; ++r) {
            int mm    = r + hi16 * 8;
            int patch = g * 16 + mm;               // 0..143
            float v = acc[r] * scale
                    + pos[patch * 256 + col]
                    + temp[f * 256 + col];
            out[((size_t)frame * 145 + 1 + patch) * 256 + col] = v;
        }
    }
}

extern "C" void kernel_launch(void* const* d_in, const int* in_sizes, int n_in,
                              void* d_out, int out_size, void* d_ws, size_t ws_size,
                              hipStream_t stream) {
    const float* x    = (const float*)d_in[0];   // [2,20,1,240,240]
    const float* emb  = (const float*)d_in[1];   // [256,256]
    const float* pos  = (const float*)d_in[2];   // [144,256]
    const float* temp = (const float*)d_in[3];   // [20,256]
    float* out = (float*)d_out;                  // [2,20,145,256]

    _Float16* Bhi = (_Float16*)d_ws;             // 128 KB
    _Float16* Blo = Bhi + NB_HALFS;              // 128 KB

    prep_emb_kernel<<<NB_HALFS / 256, 256, 0, stream>>>(emb, Bhi, Blo);
    cls_fill_kernel<<<NFRAMES, 256, 0, stream>>>(temp, out);
    video_embed_kernel<<<NFRAMES * GROUPS, 256, 0, stream>>>(x, Bhi, Blo, pos, temp, out);
}
